// MambaLayer_59785944761165
// MI455X (gfx1250) — matrix operand order repaired and checked by reading, not compiled, not run on hardware
//
#include <hip/hip_runtime.h>
#include <hip/hip_bf16.h>
#include <math.h>

typedef _Float16 v8h  __attribute__((ext_vector_type(8)));
typedef _Float16 v16h __attribute__((ext_vector_type(16)));
typedef float    v8f  __attribute__((ext_vector_type(8)));

constexpr int NB   = 2;      // batch
constexpr int NC   = 96;     // DIM
constexpr int NL   = 2048;   // L = 8*16*16
constexpr int DI   = 192;    // D_INNER
constexpr int NDIR = 3;
constexpr int E2   = 384;    // 2*D_INNER
constexpr int NEPAD= 48;     // x_proj out padded 38 -> 48
constexpr int DST  = 16;     // D_STATE
constexpr int DTR  = 6;      // DT_RANK
constexpr int NROW = NB*NL;        // 4096 rows for in/out proj
constexpr int KROW = NDIR*NB*NL;   // 12288 rows for x_proj/dt
constexpr int NSL  = 8;            // nslices
constexpr int LSL  = NL/NSL;       // 256
constexpr int NCHK = 16;           // scan chunks
constexpr int LCHK = NL/NCHK;      // 128

// direction-local index l -> original sequence position
__device__ __forceinline__ int dperm(int k, int l) {
  if (k == 0) return l;
  if (k == 1) return NL - 1 - l;
  // k==2: l = j*NSL + i  maps from original i*LSL + j
  return (l & (NSL-1)) * LSL + (l >> 3);
}

// ---------------- LayerNorm over C=96, one wave per (b,l) row -------------
__global__ void k_ln(const float* __restrict__ x, const float* __restrict__ w,
                     const float* __restrict__ bia, _Float16* __restrict__ xn) {
  int row  = blockIdx.x * (blockDim.x >> 5) + (threadIdx.x >> 5);
  int lane = threadIdx.x & 31;
  if (row >= NROW) return;
  int bb = row / NL, l = row % NL;
  const float* xp = x + (size_t)bb * NC * NL + l;      // channel stride NL
  float v0 = xp[(lane     ) * NL];
  float v1 = xp[(lane + 32) * NL];
  float v2 = xp[(lane + 64) * NL];
  float s = v0 + v1 + v2;
  #pragma unroll
  for (int off = 16; off; off >>= 1) s += __shfl_xor(s, off, 32);
  float mu = s * (1.0f / NC);
  float d0 = v0 - mu, d1 = v1 - mu, d2 = v2 - mu;
  float q = d0*d0 + d1*d1 + d2*d2;
  #pragma unroll
  for (int off = 16; off; off >>= 1) q += __shfl_xor(q, off, 32);
  float rs = rsqrtf(q * (1.0f / NC) + 1e-5f);
  _Float16* o = xn + (size_t)row * NC;
  o[lane     ] = (_Float16)(d0 * rs * w[lane     ] + bia[lane     ]);
  o[lane + 32] = (_Float16)(d1 * rs * w[lane + 32] + bia[lane + 32]);
  o[lane + 64] = (_Float16)(d2 * rs * w[lane + 64] + bia[lane + 64]);
}

// ------------- pack f32 weights [N][K] -> f16 [Npad][K], zero pad ---------
__global__ void k_pack_w(const float* __restrict__ src, _Float16* __restrict__ dst,
                         int N, int K, int Npad) {
  int i = blockIdx.x * blockDim.x + threadIdx.x;
  if (i >= Npad * K) return;
  int n = i / K, k = i % K;
  dst[i] = (n < N) ? (_Float16)src[n * K + k] : (_Float16)0.0f;
}

// ------------- WMMA GEMM: C[M][N] = A[M][K](f16) * Bpk[N][K](f16)^T -------
// one wave per 16x16 tile; K multiple of 32, M,N multiples of 16
__global__ void k_wmma_gemm(const _Float16* __restrict__ A,
                            const _Float16* __restrict__ Bp,
                            float* __restrict__ C, int M, int N, int K) {
  int wave = (blockIdx.x * blockDim.x + threadIdx.x) >> 5;
  int lane = threadIdx.x & 31;
  int ntn  = N >> 4;
  int tm = wave / ntn, tn = wave % ntn;
  if (tm * 16 >= M) return;                 // wave-uniform exit
  int hi = lane >> 4, lo = lane & 15;
  // A fragment: row = tm*16+lo ; contiguous K range [hi*16 .. hi*16+15]
  const _Float16* arow = A  + (size_t)(tm * 16 + lo) * K + hi * 16;
  // B fragment: col = tn*16+lo ; contiguous K range [hi*16 .. hi*16+15]
  const _Float16* brow = Bp + (size_t)(tn * 16 + lo) * K + hi * 16;
  v8f acc = {};
  for (int k0 = 0; k0 < K; k0 += 32) {
    v8h alo = *(const v8h*)(arow + k0);
    v8h ahi = *(const v8h*)(arow + k0 + 8);
    v16h af = __builtin_shufflevector(alo, ahi,
              0,1,2,3,4,5,6,7,8,9,10,11,12,13,14,15);
    v16h bf = *(const v16h*)(brow + k0);
    acc = __builtin_amdgcn_wmma_f32_16x16x32_f16(
              false, af, false, bf, (short)0, acc, false, false);
  }
  // D layout: VGPR r -> row tm*16 + r + 8*hi, col tn*16 + lo
  float* crow = C + (size_t)(tm * 16 + hi * 8) * N + tn * 16 + lo;
  #pragma unroll
  for (int r = 0; r < 8; ++r) crow[(size_t)r * N] = acc[r];
}

// ------- direction gather + causal depthwise conv(4) + SiLU --------------
// grid = NDIR*NB*NL blocks, 192 threads (d)
__global__ void k_conv(const float* __restrict__ xz, const float* __restrict__ cw,
                       const float* __restrict__ cb, float* __restrict__ xc32,
                       _Float16* __restrict__ xc16) {
  int blk = blockIdx.x;
  int d = threadIdx.x;
  int l  = blk % NL;
  int kb = blk / NL;
  int bb = kb % NB;
  int k  = kb / NB;
  float acc = cb[k * DI + d];
  #pragma unroll
  for (int j = 0; j < 4; ++j) {
    int ls = l - 3 + j;
    if (ls >= 0) {
      int orig = dperm(k, ls);
      acc += cw[(k * DI + d) * 4 + j] * xz[(size_t)(bb * NL + orig) * E2 + d];
    }
  }
  float sv = acc / (1.0f + __expf(-acc));      // SiLU
  size_t idx = (size_t)blk * DI + d;           // [k][b][l][d]
  xc32[idx] = sv;
  xc16[idx] = (_Float16)sv;
}

// ---------------- dt = softplus(x_dbl[:, :6] @ dt_proj_w^T + b) ----------
__global__ void k_dt(const float* __restrict__ xdbl, const float* __restrict__ dtw,
                     const float* __restrict__ dtb, float* __restrict__ dt) {
  int idx = blockIdx.x * blockDim.x + threadIdx.x;
  if (idx >= KROW * DI) return;
  int row = idx / DI, d = idx % DI;
  const float* xr = xdbl + (size_t)row * NEPAD;
  float a = dtb[d];
  #pragma unroll
  for (int r = 0; r < DTR; ++r) a += xr[r] * dtw[d * DTR + r];
  dt[idx] = (a > 20.0f) ? a : log1pf(__expf(a));
}

// ---- scan pass 1: per-chunk local scan -> (S = sum dt, G = local end h) --
__global__ void k_scan_chunk(const float* __restrict__ dt, const float* __restrict__ xc32,
                             const float* __restrict__ xdbl, const float* __restrict__ Alog,
                             float* __restrict__ csumS, float* __restrict__ csumG) {
  int kb = blockIdx.x / NCHK;
  int ch = blockIdx.x % NCHK;
  int d  = threadIdx.x;
  int k  = kb / NB;
  float A[DST], h[DST];
  #pragma unroll
  for (int n = 0; n < DST; ++n) {
    A[n] = -__expf(Alog[(size_t)(k * DI + d) * DST + n]);
    h[n] = 0.0f;
  }
  float S = 0.0f;
  int rowbase = kb * NL + ch * LCHK;
  for (int l = 0; l < LCHK; ++l) {
    int row = rowbase + l;                           // block-uniform -> s_load
    const float* xd = xdbl + (size_t)row * NEPAD;
    float dtv = dt  [(size_t)row * DI + d];
    float u   = xc32[(size_t)row * DI + d];
    S += dtv;
    #pragma unroll
    for (int n = 0; n < DST; ++n) {
      float Bm = xd[DTR + n];
      h[n] = __expf(dtv * A[n]) * h[n] + dtv * Bm * u;
    }
  }
  size_t cidx = (size_t)(kb * NCHK + ch) * DI + d;
  csumS[cidx] = S;
  #pragma unroll
  for (int n = 0; n < DST; ++n) csumG[cidx * DST + n] = h[n];
}

// ---- scan pass 2: serial exclusive prefix over chunks (tiny) ------------
__global__ void k_scan_prefix(const float* __restrict__ Alog, const float* __restrict__ csumS,
                              const float* __restrict__ csumG, float* __restrict__ hpre) {
  int kb = blockIdx.x;
  int d  = threadIdx.x;
  int k  = kb / NB;
  float A[DST], h[DST];
  #pragma unroll
  for (int n = 0; n < DST; ++n) {
    A[n] = -__expf(Alog[(size_t)(k * DI + d) * DST + n]);
    h[n] = 0.0f;
  }
  for (int ch = 0; ch < NCHK; ++ch) {
    size_t cidx = (size_t)(kb * NCHK + ch) * DI + d;
    #pragma unroll
    for (int n = 0; n < DST; ++n) hpre[cidx * DST + n] = h[n];   // exclusive
    float S = csumS[cidx];
    #pragma unroll
    for (int n = 0; n < DST; ++n)
      h[n] = __expf(A[n] * S) * h[n] + csumG[cidx * DST + n];
  }
}

// ---- scan pass 3: re-scan chunk from prefix state, gate + skip, write y --
__global__ void k_scan_out(const float* __restrict__ dt, const float* __restrict__ xc32,
                           const float* __restrict__ xdbl, const float* __restrict__ xz,
                           const float* __restrict__ Alog, const float* __restrict__ Dp,
                           const float* __restrict__ hpre, float* __restrict__ ybuf) {
  int kb = blockIdx.x / NCHK;
  int ch = blockIdx.x % NCHK;
  int d  = threadIdx.x;
  int bb = kb % NB;
  int k  = kb / NB;
  float A[DST], h[DST];
  size_t cidx = (size_t)(kb * NCHK + ch) * DI + d;
  #pragma unroll
  for (int n = 0; n < DST; ++n) {
    A[n] = -__expf(Alog[(size_t)(k * DI + d) * DST + n]);
    h[n] = hpre[cidx * DST + n];
  }
  float Dd = Dp[k * DI + d];
  int rowbase = kb * NL + ch * LCHK;
  for (int l = 0; l < LCHK; ++l) {
    int row = rowbase + l;
    const float* xd = xdbl + (size_t)row * NEPAD;
    float dtv = dt  [(size_t)row * DI + d];
    float u   = xc32[(size_t)row * DI + d];
    float y = 0.0f;
    #pragma unroll
    for (int n = 0; n < DST; ++n) {
      float Bm = xd[DTR + n];
      float Cm = xd[DTR + DST + n];
      h[n] = __expf(dtv * A[n]) * h[n] + dtv * Bm * u;
      y += h[n] * Cm;
    }
    int orig = dperm(k, ch * LCHK + l);
    float z = xz[(size_t)(bb * NL + orig) * E2 + DI + d];
    float yv = (y + Dd * u) * (z / (1.0f + __expf(-z)));
    ybuf[(size_t)row * DI + d] = yv;
  }
}

// ---- merge 3 directions back to original ordering, cvt to f16 -----------
__global__ void k_combine(const float* __restrict__ ybuf, _Float16* __restrict__ yc) {
  int idx = blockIdx.x * blockDim.x + threadIdx.x;
  if (idx >= NROW * DI) return;
  int d  = idx % DI;
  int bp = idx / DI;
  int p  = bp % NL;
  int bb = bp / NL;
  int l1 = NL - 1 - p;
  int l2 = (p % LSL) * NSL + (p / LSL);      // inverse of slice shuffle
  float v = ybuf[(size_t)((0 * NB + bb) * NL + p ) * DI + d]
          + ybuf[(size_t)((1 * NB + bb) * NL + l1) * DI + d]
          + ybuf[(size_t)((2 * NB + bb) * NL + l2) * DI + d];
  yc[idx] = (_Float16)v;
}

// ---- residual add into o[(b,l)][c] --------------------------------------
__global__ void k_resid(float* __restrict__ o, const float* __restrict__ x) {
  int idx = blockIdx.x * blockDim.x + threadIdx.x;
  if (idx >= NROW * NC) return;
  int c = idx % NC, r = idx / NC;
  int l = r % NL, bb = r / NL;
  o[idx] += x[(size_t)(bb * NC + c) * NL + l];
}

// ---- per-channel BN stats over B*L --------------------------------------
__global__ void k_bnstat(const float* __restrict__ o, float* __restrict__ stat) {
  __shared__ float sS[256], sQ[256];
  int c = blockIdx.x, t = threadIdx.x;
  float s = 0.0f, q = 0.0f;
  for (int r = t; r < NROW; r += 256) {
    float v = o[(size_t)r * NC + c];
    s += v; q += v * v;
  }
  sS[t] = s; sQ[t] = q; __syncthreads();
  for (int off = 128; off; off >>= 1) {
    if (t < off) { sS[t] += sS[t + off]; sQ[t] += sQ[t + off]; }
    __syncthreads();
  }
  if (t == 0) {
    float m = sS[0] * (1.0f / NROW);
    stat[c]      = m;
    stat[NC + c] = sQ[0] * (1.0f / NROW) - m * m;
  }
}

// ---- BN apply, write output in (B,C,D,H,W) order ------------------------
__global__ void k_bnout(const float* __restrict__ o, const float* __restrict__ stat,
                        const float* __restrict__ g, const float* __restrict__ be,
                        float* __restrict__ out) {
  int idx = blockIdx.x * blockDim.x + threadIdx.x;
  if (idx >= NB * NC * NL) return;
  int l  = idx % NL;
  int c  = (idx / NL) % NC;
  int bb = idx / (NC * NL);
  float m = stat[c], v = stat[NC + c];
  out[idx] = g[c] * (o[(size_t)(bb * NL + l) * NC + c] - m) * rsqrtf(v + 1e-5f) + be[c];
}

extern "C" void kernel_launch(void* const* d_in, const int* in_sizes, int n_in,
                              void* d_out, int out_size, void* d_ws, size_t ws_size,
                              hipStream_t stream) {
  const float* x        = (const float*)d_in[0];
  const float* ln_w     = (const float*)d_in[1];
  const float* ln_b     = (const float*)d_in[2];
  const float* in_proj  = (const float*)d_in[3];
  const float* conv_w   = (const float*)d_in[4];
  const float* conv_b   = (const float*)d_in[5];
  const float* x_proj   = (const float*)d_in[6];
  const float* dt_w     = (const float*)d_in[7];
  const float* dt_b     = (const float*)d_in[8];
  const float* A_log    = (const float*)d_in[9];
  const float* Dp       = (const float*)d_in[10];
  const float* out_proj = (const float*)d_in[11];
  const float* bn_g     = (const float*)d_in[12];
  const float* bn_b     = (const float*)d_in[13];
  float* out = (float*)d_out;

  char* w = (char*)d_ws;
  size_t off = 0;
  auto alloc = [&](size_t bytes) {
    void* p = w + off;
    off = (off + bytes + 255) & ~(size_t)255;
    return p;
  };
  _Float16* xn_f16  = (_Float16*)alloc((size_t)NROW * NC * 2);
  _Float16* w_in16  = (_Float16*)alloc((size_t)E2 * NC * 2);
  _Float16* w_xp16  = (_Float16*)alloc((size_t)NEPAD * DI * 2);
  _Float16* w_op16  = (_Float16*)alloc((size_t)NC * DI * 2);
  float*    xz      = (float*)   alloc((size_t)NROW * E2 * 4);
  _Float16* xc16    = (_Float16*)alloc((size_t)KROW * DI * 2);
  float*    xc32    = (float*)   alloc((size_t)KROW * DI * 4);
  float*    xdbl    = (float*)   alloc((size_t)KROW * NEPAD * 4);
  float*    dtb     = (float*)   alloc((size_t)KROW * DI * 4);
  float*    csumS   = (float*)   alloc((size_t)NDIR * NB * NCHK * DI * 4);
  float*    csumG   = (float*)   alloc((size_t)NDIR * NB * NCHK * DI * DST * 4);
  float*    hpre    = (float*)   alloc((size_t)NDIR * NB * NCHK * DI * DST * 4);
  float*    ybuf    = (float*)   alloc((size_t)KROW * DI * 4);
  _Float16* yc16    = (_Float16*)alloc((size_t)NROW * DI * 2);
  float*    obuf    = (float*)   alloc((size_t)NROW * NC * 4);
  float*    stat    = (float*)   alloc((size_t)2 * NC * 4);
  (void)in_sizes; (void)n_in; (void)out_size; (void)ws_size;

  // 1) LayerNorm -> f16 activations
  k_ln<<<NROW / 4, 128, 0, stream>>>(x, ln_w, ln_b, xn_f16);

  // 2) pack weights to f16 [N][K]
  k_pack_w<<<(E2 * NC + 255) / 256, 256, 0, stream>>>(in_proj, w_in16, E2, NC, E2);
  k_pack_w<<<(NEPAD * DI + 255) / 256, 256, 0, stream>>>(x_proj, w_xp16, DTR + 2 * DST, DI, NEPAD);
  k_pack_w<<<(NC * DI + 255) / 256, 256, 0, stream>>>(out_proj, w_op16, NC, DI, NC);

  // 3) in_proj GEMM: 4096 x 96 -> 384   (WMMA)
  {
    int tiles = (NROW / 16) * (E2 / 16);
    k_wmma_gemm<<<(tiles + 7) / 8, 256, 0, stream>>>(xn_f16, w_in16, xz, NROW, E2, NC);
  }

  // 4) direction gather + conv + SiLU
  k_conv<<<NDIR * NB * NL, DI, 0, stream>>>(xz, conv_w, conv_b, xc32, xc16);

  // 5) x_proj GEMM: 12288 x 192 -> 48   (WMMA)
  {
    int tiles = (KROW / 16) * (NEPAD / 16);
    k_wmma_gemm<<<(tiles + 7) / 8, 256, 0, stream>>>(xc16, w_xp16, xdbl, KROW, NEPAD, DI);
  }

  // 6) dt projection + softplus
  k_dt<<<(KROW * DI + 255) / 256, 256, 0, stream>>>(xdbl, dt_w, dt_b, dtb);

  // 7) chunked selective scan (3 passes)
  k_scan_chunk <<<NDIR * NB * NCHK, DI, 0, stream>>>(dtb, xc32, xdbl, A_log, csumS, csumG);
  k_scan_prefix<<<NDIR * NB,        DI, 0, stream>>>(A_log, csumS, csumG, hpre);
  k_scan_out   <<<NDIR * NB * NCHK, DI, 0, stream>>>(dtb, xc32, xdbl, xz, A_log, Dp, hpre, ybuf);

  // 8) merge directions -> f16
  k_combine<<<(NROW * DI + 255) / 256, 256, 0, stream>>>(ybuf, yc16);

  // 9) out_proj GEMM: 4096 x 192 -> 96  (WMMA)
  {
    int tiles = (NROW / 16) * (NC / 16);
    k_wmma_gemm<<<(tiles + 7) / 8, 256, 0, stream>>>(yc16, w_op16, obuf, NROW, NC, DI);
  }

  // 10) residual + BatchNorm
  k_resid <<<(NROW * NC + 255) / 256, 256, 0, stream>>>(obuf, x);
  k_bnstat<<<NC, 256, 0, stream>>>(obuf, stat);
  k_bnout <<<(NB * NC * NL + 255) / 256, 256, 0, stream>>>(obuf, stat, bn_g, bn_b, out);
}
